// LocallyConnectedLayer_36215164240656
// MI455X (gfx1250) — matrix operand order repaired and checked by reading, not compiled, not run on hardware
//
#include <hip/hip_runtime.h>

// ---------------------------------------------------------------------------
// y = x @ (W * M)^T + b   on gfx1250 via v_wmma_f32_16x16x32_bf16
//   pass 1: convert x (fp32 -> bf16), and W*M -> bf16 into d_ws
//   pass 2: 128x128-tile GEMM, TK=64, LDS double-buffered,
//           global->LDS via GLOBAL_LOAD_ASYNC_TO_LDS_B128 (ASYNCcnt),
//           8 waves * 16 WMMA per K-step, one barrier per K-step
// ---------------------------------------------------------------------------

#ifndef USE_ASYNC_LDS
#define USE_ASYNC_LDS 1   // flip to 0 if the async mnemonic is rejected
#endif

typedef __bf16 v16bf __attribute__((ext_vector_type(16)));
typedef __bf16 v8bf  __attribute__((ext_vector_type(8)));
typedef __bf16 v4bf  __attribute__((ext_vector_type(4)));
typedef float  v8f   __attribute__((ext_vector_type(8)));

#define NDIM   4096
#define TM     128
#define TN     128
#define TK     64
#define LDT    72            // padded LDS row stride: 36 dwords -> conflict-free b128
#define NKSTEP (NDIM / TK)   // 64

// ---------------- prep: fp32 -> bf16 conversions ---------------------------

__global__ __launch_bounds__(256) void cvt_x_kernel(const float* __restrict__ x,
                                                    __bf16* __restrict__ o) {
  const size_t i = ((size_t)blockIdx.x * 256 + threadIdx.x) * 4;
  const float4 v = *(const float4*)(x + i);
  v4bf r;
  r[0] = (__bf16)v.x; r[1] = (__bf16)v.y; r[2] = (__bf16)v.z; r[3] = (__bf16)v.w;
  *(v4bf*)(o + i) = r;
}

__global__ __launch_bounds__(256) void cvt_maskw_kernel(const float* __restrict__ w,
                                                        const float* __restrict__ f,
                                                        __bf16* __restrict__ o) {
  const size_t i = ((size_t)blockIdx.x * 256 + threadIdx.x) * 4;
  const float4 wv = *(const float4*)(w + i);
  const float4 fv = *(const float4*)(f + i);
  v4bf r;
  r[0] = (__bf16)(wv.x * fv.x); r[1] = (__bf16)(wv.y * fv.y);
  r[2] = (__bf16)(wv.z * fv.z); r[3] = (__bf16)(wv.w * fv.w);
  *(v4bf*)(o + i) = r;
}

// ---------------- async-copy helpers ---------------------------------------

#if USE_ASYNC_LDS
__device__ __forceinline__ void async_b128(unsigned lds_byte_off, const __bf16* g) {
  // per-lane: LDS[lds_byte_off] <= 16B from g ; tracked by ASYNCcnt
  asm volatile("global_load_async_to_lds_b128 %0, %1, off"
               :: "v"(lds_byte_off), "v"(g)
               : "memory");
}
__device__ __forceinline__ void wait_async0() {
  asm volatile("s_wait_asynccnt 0" ::: "memory");
}
#endif

// ---------------- main GEMM ------------------------------------------------

__global__ __launch_bounds__(256) void lc_gemm_bf16(const __bf16* __restrict__ X,
                                                    const __bf16* __restrict__ W,
                                                    const float*  __restrict__ bias,
                                                    float*        __restrict__ Y) {
  __shared__ __bf16 sA[2][TM * LDT];   // 2 * 128 * 72 * 2B = 36 KB
  __shared__ __bf16 sB[2][TN * LDT];   // 36 KB

  const int tid  = threadIdx.x;
  const int wave = tid >> 5;
  const int lane = tid & 31;
  const int half = lane >> 4;   // WMMA lane-half (K split)
  const int r    = lane & 15;   // row-in-tile (A) / col-in-tile (B,C)

  const int m0 = blockIdx.y * TM;
  const int n0 = blockIdx.x * TN;
  const int wm = (wave >> 1) * 32;  // wave sub-tile: 32 rows
  const int wn = (wave & 1) * 64;   //                64 cols

  // panel fill: 128 rows x 64 bf16 = 1024 x 16B chunks -> 4 chunks/thread/panel
  const int crow = tid >> 3;           // 0..31
  const int ccol = (tid & 7) * 8;      // 0,8,...,56
  size_t   gA[4], gB[4];
  int      lofs[4];
  #pragma unroll
  for (int i = 0; i < 4; ++i) {
    const int rr = crow + 32 * i;
    gA[i]   = (size_t)(m0 + rr) * NDIM + ccol;
    gB[i]   = (size_t)(n0 + rr) * NDIM + ccol;
    lofs[i] = (rr * LDT + ccol) * 2;   // byte offset within a panel
  }
  const unsigned baseA[2] = { (unsigned)(size_t)(void*)&sA[0][0],
                              (unsigned)(size_t)(void*)&sA[1][0] };
  const unsigned baseB[2] = { (unsigned)(size_t)(void*)&sB[0][0],
                              (unsigned)(size_t)(void*)&sB[1][0] };

  v8f acc[2][4] = {};   // 2 M-tiles x 4 N-tiles of 16x16 f32

#if USE_ASYNC_LDS
  // prologue: launch async fill of buffer 0
  #pragma unroll
  for (int i = 0; i < 4; ++i) {
    async_b128(baseA[0] + lofs[i], X + gA[i]);
    async_b128(baseB[0] + lofs[i], W + gB[i]);
  }

  for (int ks = 0; ks < NKSTEP; ++ks) {
    const int cur = ks & 1;
    wait_async0();        // own loads into buf[cur] have landed
    __syncthreads();      // everyone's landed; everyone done reading buf[cur^1]

    if (ks + 1 < NKSTEP) {  // overlap next panel fill with compute
      const int k1 = (ks + 1) * TK;
      #pragma unroll
      for (int i = 0; i < 4; ++i) {
        async_b128(baseA[cur ^ 1] + lofs[i], X + gA[i] + k1);
        async_b128(baseB[cur ^ 1] + lofs[i], W + gB[i] + k1);
      }
    }
#else
  // fallback: synchronous reg-staged double buffering
  {
    int4 ra[4], rb[4];
    #pragma unroll
    for (int i = 0; i < 4; ++i) {
      ra[i] = *(const int4*)(X + gA[i]);
      rb[i] = *(const int4*)(W + gB[i]);
    }
    #pragma unroll
    for (int i = 0; i < 4; ++i) {
      *(int4*)((char*)&sA[0][0] + lofs[i]) = ra[i];
      *(int4*)((char*)&sB[0][0] + lofs[i]) = rb[i];
    }
  }
  __syncthreads();

  for (int ks = 0; ks < NKSTEP; ++ks) {
    const int cur = ks & 1;
    int4 ra[4], rb[4];
    if (ks + 1 < NKSTEP) {
      const int k1 = (ks + 1) * TK;
      #pragma unroll
      for (int i = 0; i < 4; ++i) {
        ra[i] = *(const int4*)(X + gA[i] + k1);
        rb[i] = *(const int4*)(W + gB[i] + k1);
      }
    }
#endif

    // ---- compute on buf[cur]: 2 K-slices of 32, 8 WMMAs each ----
    #pragma unroll
    for (int kk = 0; kk < 2; ++kk) {
      v16bf aF[2];
      #pragma unroll
      for (int t = 0; t < 2; ++t) {
        const __bf16* p = &sA[cur][(wm + t * 16 + r) * LDT + kk * 32 + half * 8];
        v8bf lo = *(const v8bf*)p;
        v8bf hi = *(const v8bf*)(p + 16);
        aF[t] = __builtin_shufflevector(lo, hi,
                0, 1, 2, 3, 4, 5, 6, 7, 8, 9, 10, 11, 12, 13, 14, 15);
      }
      v16bf bF[4];
      #pragma unroll
      for (int u = 0; u < 4; ++u) {
        const __bf16* p = &sB[cur][(wn + u * 16 + r) * LDT + kk * 32 + half * 8];
        v8bf lo = *(const v8bf*)p;
        v8bf hi = *(const v8bf*)(p + 16);
        bF[u] = __builtin_shufflevector(lo, hi,
                0, 1, 2, 3, 4, 5, 6, 7, 8, 9, 10, 11, 12, 13, 14, 15);
      }
      #pragma unroll
      for (int t = 0; t < 2; ++t)
        #pragma unroll
        for (int u = 0; u < 4; ++u)
          acc[t][u] = __builtin_amdgcn_wmma_f32_16x16x32_bf16(
              false, aF[t], false, bF[u], (short)0, acc[t][u], false, false);
    }

#if USE_ASYNC_LDS
  }
#else
    if (ks + 1 < NKSTEP) {
      const int nxt = cur ^ 1;
      #pragma unroll
      for (int i = 0; i < 4; ++i) {
        *(int4*)((char*)&sA[nxt][0] + lofs[i]) = ra[i];
        *(int4*)((char*)&sB[nxt][0] + lofs[i]) = rb[i];
      }
      __syncthreads();
    }
  }
#endif

  // epilogue: C/D layout -> VGPR p holds (M = half*8 + p, N = r)
  #pragma unroll
  for (int u = 0; u < 4; ++u) {
    const int   gcol = n0 + wn + u * 16 + r;
    const float bv   = bias[gcol];
    #pragma unroll
    for (int t = 0; t < 2; ++t) {
      const int growBase = m0 + wm + t * 16 + half * 8;
      #pragma unroll
      for (int p = 0; p < 8; ++p)
        Y[(size_t)(growBase + p) * NDIM + gcol] = acc[t][u][p] + bv;
    }
  }
}

// ---------------- launch ---------------------------------------------------

extern "C" void kernel_launch(void* const* d_in, const int* in_sizes, int n_in,
                              void* d_out, int out_size, void* d_ws, size_t ws_size,
                              hipStream_t stream) {
  const float* x    = (const float*)d_in[0];   // [4096, 4096]
  const float* w    = (const float*)d_in[1];   // [4096, 4096]
  const float* bias = (const float*)d_in[2];   // [4096]
  const float* filt = (const float*)d_in[3];   // [4096, 4096]
  float* y = (float*)d_out;

  __bf16* xb = (__bf16*)d_ws;                       // 32 MB
  __bf16* wb = xb + (size_t)NDIM * NDIM;            // 32 MB

  const size_t nElem = (size_t)NDIM * NDIM;
  const unsigned cvtBlocks = (unsigned)(nElem / 4 / 256);  // 16384

  cvt_x_kernel    <<<cvtBlocks, 256, 0, stream>>>(x, xb);
  cvt_maskw_kernel<<<cvtBlocks, 256, 0, stream>>>(w, filt, wb);

  dim3 grid(NDIM / TN, NDIM / TM);  // 32 x 32
  lc_gemm_bf16<<<grid, 256, 0, stream>>>(xb, wb, bias, y);
}
